// Linear_15719580304084
// MI455X (gfx1250) — 8-bit matrix operand order repaired and checked by reading, compiled, not run on hardware
//
#include <hip/hip_runtime.h>
#include <stdint.h>


typedef int v8i __attribute__((ext_vector_type(8)));
typedef int v4i __attribute__((ext_vector_type(4)));

#define M_TOTAL 524288
#define K_DIM   256
#define N_DIM   256
#define MTILES  (M_TOTAL / 16)

__device__ __forceinline__ int pack4(int a, int b, int c, int d) {
  return (a & 0xFF) | ((b & 0xFF) << 8) | ((c & 0xFF) << 16) | ((d & 0xFF) << 24);
}

__global__ __launch_bounds__(256) void qlinear_i8_wmma_kernel(
    const int*   __restrict__ x_int,
    const int*   __restrict__ sumin_int,
    const float* __restrict__ weight,
    const float* __restrict__ bias,
    const float* __restrict__ input_scale,
    const float* __restrict__ weight_scale,
    const float* __restrict__ bias_scale,
    const float* __restrict__ sumin_scale,
    const float* __restrict__ out_scale,
    float*       __restrict__ out)
{
  __shared__ int ldsW[N_DIM * (K_DIM / 4)];
  __shared__ int ldsA[16 * (K_DIM / 4)];

  const int tid  = threadIdx.x;
  const int lane = tid & 31;
  const int wave = tid >> 5;
  const int half = lane >> 4;
  const int l15  = lane & 15;

  {
    const float4* wrow = (const float4*)(weight + (size_t)tid * K_DIM);
    int* drow = ldsW + tid * (K_DIM / 4);
#pragma unroll
    for (int j = 0; j < K_DIM / 4; ++j) {
      float4 f = wrow[j];
      drow[j] = pack4((int)f.x, (int)f.y, (int)f.z, (int)f.w);
    }
  }
  __syncthreads();

  const float in_s    = input_scale[0];
  const float sum_s   = sumin_scale[0];
  const float inv_out = 1.0f / out_scale[0];

  v8i  bfrag[2][4];
  float dq[2], bq[2];
  int   ncol[2];
#pragma unroll
  for (int s = 0; s < 2; ++s) {
    const int col = (wave * 2 + s) * 16 + l15;
    ncol[s] = col;
    dq[s]   = in_s * weight_scale[col];
    bq[s]   = bias[col] * bias_scale[col];
#pragma unroll
    for (int kc = 0; kc < 4; ++kc) {
#pragma unroll
      for (int v = 0; v < 8; ++v) {
        const int koff = (v >> 2) * 8 + half * 4 + (v & 3);
        bfrag[s][kc][v] = ldsW[col * (K_DIM / 4) + kc * 16 + koff];
      }
    }
  }

  for (int mt = blockIdx.x; mt < MTILES; mt += gridDim.x) {
    {
      const v4i* xsrc = (const v4i*)(x_int + (size_t)mt * 16 * K_DIM + tid * 16);
      int* adst = ldsA + tid * 4;
#pragma unroll
      for (int j = 0; j < 4; ++j) {
        v4i xi = xsrc[j];
        adst[j] = pack4(xi[0], xi[1], xi[2], xi[3]);
      }
      const int mnext = mt + gridDim.x;
      if (mnext < MTILES)
        __builtin_prefetch(x_int + (size_t)mnext * 16 * K_DIM + tid * 16, 0, 1);
    }
    __syncthreads();

    v8i afrag[4];
#pragma unroll
    for (int kc = 0; kc < 4; ++kc) {
#pragma unroll
      for (int v = 0; v < 8; ++v) {
        const int koff = (v >> 2) * 8 + half * 4 + (v & 3);
        afrag[kc][v] = ldsA[l15 * (K_DIM / 4) + kc * 16 + koff];
      }
    }

#pragma unroll
    for (int s = 0; s < 2; ++s) {
      v8i acc = {};
#pragma unroll
      for (int kc = 0; kc < 4; ++kc) {
        acc = __builtin_amdgcn_wmma_i32_16x16x64_iu8(
            true, afrag[kc], true, bfrag[s][kc], acc, false, false);
      }

#pragma unroll
      for (int g = 0; g < 8; ++g) {
        const long long row = (long long)mt * 16 + g + half * 8;
        const long long idx = row * N_DIM + ncol[s];
        float f = (float)acc[g] * dq[s] + bq[s] + (float)sumin_int[idx] * sum_s;
        float q = rintf(f * inv_out);
        q = fminf(fmaxf(q, -128.0f), 127.0f);
        out[idx] = q;
      }
    }
    __syncthreads();
  }
}

extern "C" void kernel_launch(void* const* d_in, const int* in_sizes, int n_in,
                              void* d_out, int out_size, void* d_ws, size_t ws_size,
                              hipStream_t stream) {
  (void)in_sizes; (void)n_in; (void)out_size; (void)d_ws; (void)ws_size;

  const int*   x_int        = (const int*)d_in[0];
  const int*   sumin_int    = (const int*)d_in[1];
  const float* weight       = (const float*)d_in[2];
  const float* bias         = (const float*)d_in[3];
  const float* input_scale  = (const float*)d_in[4];
  const float* weight_scale = (const float*)d_in[5];
  const float* bias_scale   = (const float*)d_in[6];
  const float* sumin_scale  = (const float*)d_in[7];
  const float* out_scale    = (const float*)d_in[8];
  float*       out          = (float*)d_out;

  dim3 grid(1024), block(256);
  qlinear_i8_wmma_kernel<<<grid, block, 0, stream>>>(
      x_int, sumin_int, weight, bias, input_scale, weight_scale,
      bias_scale, sumin_scale, out_scale, out);
}
